// SR_24429773980149
// MI455X (gfx1250) — compile-verified, matrix operand order checked by reading, not run on hardware
//
#include <hip/hip_runtime.h>
#include <hip/hip_bf16.h>
#include <cstdint>

typedef _Float16 v16h __attribute__((ext_vector_type(16)));
typedef float    v8f  __attribute__((ext_vector_type(8)));
typedef unsigned int u32x4 __attribute__((ext_vector_type(4)));
typedef int          i32x4 __attribute__((ext_vector_type(4)));
typedef int          i32x8 __attribute__((ext_vector_type(8)));

#define NTOK 2304   // 48*48
#define CCH  256
#define NB   8
#define TQ   64     // query rows per workgroup
#define TK   64     // key cols per tile
#define KSTR 272    // LDS stride for 256-wide f16 rows (544B, 16B-aligned)
#define SSTR 80     // LDS stride for 64-wide f16 P rows (160B, 16B-aligned)

// ---------------------------------------------------------------------------
// Fragment loader for V_WMMA_F32_16X16X32_F16.
// 16-bit A (16x32) layout per lane L: row/col = L%16, g = L/16.
// half-element e: e in [0,8)  -> K = kbase + 8*g + e
//                 e in [8,16) -> K = kbase + 16 + 8*g + (e-8)
// => two contiguous 16-byte loads per lane.  B (32x16) mirrors it with the
// lane selecting the column, so the same loader serves [col][k] storage.
// ---------------------------------------------------------------------------
__device__ __forceinline__ v16h frag_ld(const _Float16* base, int stride,
                                        int kbase, int lane) {
  const int row = lane & 15;
  const int g   = (lane >> 4) & 1;
  const _Float16* p = base + (size_t)row * stride + kbase + 8 * g;
  union { v16h v; uint4 q[2]; } u;
  u.q[0] = *(const uint4*)(p);
  u.q[1] = *(const uint4*)(p + 16);
  return u.v;
}

__device__ __forceinline__ v8f zero8() {
  v8f z;
#pragma unroll
  for (int i = 0; i < 8; ++i) z[i] = 0.0f;
  return z;
}

__device__ __forceinline__ v8f wmma_f16(v16h a, v16h b, v8f c) {
  return __builtin_amdgcn_wmma_f32_16x16x32_f16(false, a, false, b,
                                                (short)0, c, false, false);
}

// Issue a TDM load of a 64-row x 512B tile (row stride 512B in memory) into
// LDS with 32B padding appended per row (=> 544B LDS row stride == KSTR).
// Must be executed under a *wave-uniform* branch: TDM ignores EXEC.
__device__ __forceinline__ void tdm_load_k_tile(const _Float16* gsrc,
                                                uint32_t lds_byte_addr) {
  const uint64_t ga = (uint64_t)(uintptr_t)gsrc;
  u32x4 g0;
  g0[0] = 1u;                                   // count=1, gather off
  g0[1] = lds_byte_addr;                        // LDS dest (bytes)
  g0[2] = (uint32_t)ga;                         // global_addr[31:0]
  g0[3] = (uint32_t)(ga >> 32) | 0x80000000u;   // global_addr[56:32] | type=2
  i32x8 g1;
  g1[0] = (1 << 16)      // data_size = 2 bytes
        | (1 << 20)      // pad_enable
        | (6 << 22)      // pad_interval: 128 DWORDs (=512B) before pad
        | (7 << 25);     // pad_amount: 8 DWORDs (=32B)
  g1[1] = (int)(256u << 16);   // tensor_dim0 = 256 (low 16 bits)
  g1[2] = (int)(64u << 16);    // tensor_dim1 = 64  (low 16 bits)
  g1[3] = (int)(256u << 16);   // tile_dim0 = 256
  g1[4] = 64;                  // tile_dim1 = 64
  g1[5] = 256;                 // tensor_dim0_stride = 256 elements
  g1[6] = 0;
  g1[7] = 0;
  i32x4 gz = {0, 0, 0, 0};
#if __clang_major__ >= 23
  i32x8 gz8 = {0, 0, 0, 0, 0, 0, 0, 0};
  __builtin_amdgcn_tensor_load_to_lds(g0, g1, gz, gz, gz8, 0);
#else
  __builtin_amdgcn_tensor_load_to_lds(g0, g1, gz, gz, 0);
#endif
}

// ---------------------------------------------------------------------------
// Kernel 0: convert the six 256x256 f32 weight matrices to f16 once.
// grid = (32, 6), block = 256; each thread converts 8 consecutive elements.
// ---------------------------------------------------------------------------
__global__ __launch_bounds__(256) void wcvt_kernel(
    const float* __restrict__ w0, const float* __restrict__ w1,
    const float* __restrict__ w2, const float* __restrict__ w3,
    const float* __restrict__ w4, const float* __restrict__ w5,
    _Float16* __restrict__ wbuf) {
  const float* src;
  switch (blockIdx.y) {
    case 0: src = w0; break;
    case 1: src = w1; break;
    case 2: src = w2; break;
    case 3: src = w3; break;
    case 4: src = w4; break;
    default: src = w5; break;
  }
  _Float16* dst = wbuf + (size_t)blockIdx.y * CCH * CCH;
  const int base = blockIdx.x * 2048 + threadIdx.x * 8;
  float4 a = *(const float4*)(src + base);
  float4 b = *(const float4*)(src + base + 4);
  union { uint4 q; _Float16 h[8]; } pk;
  pk.h[0] = (_Float16)a.x; pk.h[1] = (_Float16)a.y;
  pk.h[2] = (_Float16)a.z; pk.h[3] = (_Float16)a.w;
  pk.h[4] = (_Float16)b.x; pk.h[5] = (_Float16)b.y;
  pk.h[6] = (_Float16)b.z; pk.h[7] = (_Float16)b.w;
  *(uint4*)(dst + base) = pk.q;
}

// ---------------------------------------------------------------------------
// Kernel 1: QKV projections.  grid = (36, 3, 16), block = 256.
//   y: 0=Q, 1=K^T, 2=V(transposed);  z: dir*8 + batch
// q/k buffers: f16 [dir][b][n][c];  v buffer: f16 [dir][b][c][n]
// ---------------------------------------------------------------------------
__global__ __launch_bounds__(256) void proj_kernel(
    const float* __restrict__ rgb, const float* __restrict__ t,
    const float* __restrict__ bq1, const float* __restrict__ bk1,
    const float* __restrict__ bv1, const float* __restrict__ bq2,
    const float* __restrict__ bk2, const float* __restrict__ bv2,
    const _Float16* __restrict__ wbuf,
    _Float16* __restrict__ qbuf, _Float16* __restrict__ kbuf,
    _Float16* __restrict__ vbuf) {
  __shared__ _Float16 Xs[TQ][KSTR];

  const int tid  = threadIdx.x;
  const int lane = tid & 31;
  const int w    = tid >> 5;
  const int qkv  = blockIdx.y;
  const int zz   = blockIdx.z;
  const int d    = zz >> 3;
  const int b    = zz & 7;
  const int n0   = blockIdx.x * TQ;

  const float* x = (d == 0) ? rgb : t;
  const float* y = (d == 0) ? t : rgb;
  const float *bias, *src;
  _Float16* dstb;
  if (qkv == 0)      { bias = d ? bq2 : bq1; src = x; dstb = qbuf; }
  else if (qkv == 1) { bias = d ? bk2 : bk1; src = y; dstb = kbuf; }
  else               { bias = d ? bv2 : bv1; src = y; dstb = vbuf; }
  const _Float16* Wh = wbuf + (size_t)(d * 3 + qkv) * CCH * CCH;  // [o][k] f16
  src += (size_t)b * CCH * NTOK;
  _Float16* dst = dstb + (size_t)(d * NB + b) * NTOK * CCH;

  // stage transposed input tile: Xs[n][i] = src[i][n0+n], converted to f16
  for (int u = tid; u < TQ * CCH; u += 256) {
    const int i = u >> 6;
    const int n = u & 63;
    Xs[n][i] = (_Float16)src[(size_t)i * NTOK + n0 + n];
  }
  __syncthreads();

  // each wave: 2 column blocks x 4 row blocks = 8 output fragments
#pragma unroll
  for (int fi = 0; fi < 8; ++fi) {
    const int cb  = (w << 1) + (fi & 1);  // 0..15
    const int rbp = fi >> 1;              // 0..3
    v8f acc = zero8();
#pragma unroll
    for (int kk = 0; kk < 8; ++kk) {
      v16h a  = frag_ld(&Xs[rbp * 16][0], KSTR, kk * 32, lane);
      v16h bb = frag_ld(Wh + (size_t)cb * 16 * CCH, CCH, kk * 32, lane);
      acc = wmma_f16(a, bb, acc);
    }
    const int o  = cb * 16 + (lane & 15);
    const int g  = lane >> 4;
    const float bval = bias[o];
    if (qkv < 2) {
      // [n][c] layout
#pragma unroll
      for (int r = 0; r < 8; ++r) {
        const int m = rbp * 16 + r + 8 * g;
        dst[(size_t)(n0 + m) * CCH + o] = (_Float16)(acc[r] + bval);
      }
    } else {
      // [c][n] layout: this lane's 8 rows are contiguous -> one 16B store
      union { uint4 q; _Float16 h[8]; } pk;
#pragma unroll
      for (int r = 0; r < 8; ++r) pk.h[r] = (_Float16)(acc[r] + bval);
      *(uint4*)&dst[(size_t)o * NTOK + n0 + rbp * 16 + 8 * g] = pk.q;
    }
  }
}

// ---------------------------------------------------------------------------
// Kernel 2: flash cross-attention + residuals.  grid = (36, 1, 16), block 256.
// K tiles are staged by the Tensor Data Mover into a double-buffered LDS
// tile (hardware-padded rows), overlapped with WMMA compute.
// out layout: [dir][b][c][n] (two outputs concatenated)
// ---------------------------------------------------------------------------
__global__ __launch_bounds__(256) void attn_kernel(
    const float* __restrict__ rgb, const float* __restrict__ t,
    const _Float16* __restrict__ qbuf, const _Float16* __restrict__ kbuf,
    const _Float16* __restrict__ vbuf, float* __restrict__ out) {
  __shared__ _Float16 Ks[2][TK][KSTR];  // 2 x 34816 B (TDM destination)
  __shared__ float    Sf[TQ][68];       // 17408 B
  __shared__ _Float16 Ss[TQ][SSTR];     // 10240 B
  __shared__ float    corrS[TQ];
  __shared__ float    lS[TQ];

  const int tid  = threadIdx.x;
  const int lane = tid & 31;
  const int w    = tid >> 5;
  const int rb   = w >> 1;   // row block 0..3 (16 rows each)
  const int ch   = w & 1;    // column half 0..1 (128 cols each)
  const int zz   = blockIdx.z;
  const int d    = zz >> 3;
  const int b    = zz & 7;
  const int n0   = blockIdx.x * TQ;

  const size_t boff = (size_t)(d * NB + b) * NTOK * CCH;
  const _Float16* qb = qbuf + boff;
  const _Float16* kb = kbuf + boff;
  const _Float16* vb = vbuf + boff;

  // generic LDS pointers truncate to the in-group-segment byte offset
  const uint32_t ldsK0 = (uint32_t)(uintptr_t)&Ks[0][0][0];
  const uint32_t ldsK1 = (uint32_t)(uintptr_t)&Ks[1][0][0];

  // Q A-fragments for this wave's 16 rows, all 8 K-steps (held in VGPRs)
  v16h qfrag[8];
  {
    const _Float16* qrow = qb + (size_t)(n0 + rb * 16) * CCH;
#pragma unroll
    for (int kk = 0; kk < 8; ++kk)
      qfrag[kk] = frag_ld(qrow, CCH, kk * 32, lane);
  }

  v8f oacc[8];
#pragma unroll
  for (int i = 0; i < 8; ++i) oacc[i] = zero8();

  float m_i = -3.0e38f, l_i = 0.0f;  // live in threads tid<64

  // prefetch first K tile via TDM (wave-uniform branch: TDM ignores EXEC)
  if (w == 0) tdm_load_k_tile(kb, ldsK0);

  for (int jt = 0; jt < NTOK / TK; ++jt) {
    const int j0  = jt * TK;
    const int buf = jt & 1;

    if (w == 0) {
      if (jt + 1 < NTOK / TK) {
        // issue next tile into the other buffer, then wait for current
        tdm_load_k_tile(kb + (size_t)(j0 + TK) * CCH, buf ? ldsK0 : ldsK1);
        __builtin_amdgcn_s_wait_tensorcnt(1);
      } else {
        __builtin_amdgcn_s_wait_tensorcnt(0);
      }
    }
    __syncthreads();

    // S = Q * K^T : each wave computes 2 of 16 fragments
#pragma unroll
    for (int fi = 0; fi < 2; ++fi) {
      const int cbS = ch * 2 + fi;  // column block within the 64-wide tile
      v8f acc = zero8();
#pragma unroll
      for (int kk = 0; kk < 8; ++kk) {
        v16h bfr = frag_ld(&Ks[buf][cbS * 16][0], KSTR, kk * 32, lane);
        acc = wmma_f16(qfrag[kk], bfr, acc);
      }
      const int col = cbS * 16 + (lane & 15);
      const int g   = lane >> 4;
#pragma unroll
      for (int r = 0; r < 8; ++r)
        Sf[rb * 16 + r + 8 * g][col] = acc[r];
    }
    __syncthreads();

    // online softmax: one thread per query row
    if (tid < TQ) {
      const int row = tid;
      float mx = -3.0e38f;
#pragma unroll 8
      for (int j = 0; j < TK; ++j) mx = fmaxf(mx, Sf[row][j]);
      const float mn   = fmaxf(m_i, mx);
      const float corr = __expf(m_i - mn);
      float s = 0.0f;
#pragma unroll 8
      for (int j = 0; j < TK; ++j) {
        const float p = __expf(Sf[row][j] - mn);
        s += p;
        Ss[row][j] = (_Float16)p;
      }
      l_i = l_i * corr + s;
      m_i = mn;
      corrS[row] = corr;
    }
    __syncthreads();

    // rescale running O, then O += P * V
    {
      const int g = lane >> 4;
      float cr[8];
#pragma unroll
      for (int r = 0; r < 8; ++r) cr[r] = corrS[rb * 16 + r + 8 * g];
#pragma unroll
      for (int c8 = 0; c8 < 8; ++c8)
#pragma unroll
        for (int r = 0; r < 8; ++r) oacc[c8][r] *= cr[r];
    }
    {
      v16h pa0 = frag_ld(&Ss[rb * 16][0], SSTR, 0, lane);
      v16h pa1 = frag_ld(&Ss[rb * 16][0], SSTR, 32, lane);
#pragma unroll
      for (int c8 = 0; c8 < 8; ++c8) {
        const int co = ch * 128 + c8 * 16;
        const _Float16* vrow = vb + (size_t)co * NTOK + j0;  // [c][n] major
        v16h vb0 = frag_ld(vrow, NTOK, 0, lane);
        v16h vb1 = frag_ld(vrow, NTOK, 32, lane);
        oacc[c8] = wmma_f16(pa0, vb0, oacc[c8]);
        oacc[c8] = wmma_f16(pa1, vb1, oacc[c8]);
      }
    }
    __syncthreads();
  }

  if (tid < TQ) lS[tid] = l_i;
  __syncthreads();

  // epilogue: out = x + q + O / l   (stored [c][n]-major)
  const float* xsrc = ((d == 0) ? rgb : t) + (size_t)b * CCH * NTOK;
  float* dst = out + (size_t)d * NB * CCH * NTOK + (size_t)b * CCH * NTOK;
  const int g = lane >> 4;
  float linv[8];
#pragma unroll
  for (int r = 0; r < 8; ++r) linv[r] = 1.0f / lS[rb * 16 + r + 8 * g];

#pragma unroll
  for (int c8 = 0; c8 < 8; ++c8) {
    const int o = ch * 128 + c8 * 16 + (lane & 15);
    const size_t base = (size_t)o * NTOK + n0 + rb * 16 + 8 * g;
    float4 x0 = *(const float4*)&xsrc[base];
    float4 x1 = *(const float4*)&xsrc[base + 4];
    const float xr[8] = {x0.x, x0.y, x0.z, x0.w, x1.x, x1.y, x1.z, x1.w};
    float vals[8];
#pragma unroll
    for (int r = 0; r < 8; ++r) {
      const int m = rb * 16 + r + 8 * g;
      const float qv = (float)qb[(size_t)(n0 + m) * CCH + o];
      vals[r] = xr[r] + qv + oacc[c8][r] * linv[r];
    }
    float4 s0 = {vals[0], vals[1], vals[2], vals[3]};
    float4 s1 = {vals[4], vals[5], vals[6], vals[7]};
    *(float4*)&dst[base]     = s0;
    *(float4*)&dst[base + 4] = s1;
  }
}

// ---------------------------------------------------------------------------
extern "C" void kernel_launch(void* const* d_in, const int* in_sizes, int n_in,
                              void* d_out, int out_size, void* d_ws,
                              size_t ws_size, hipStream_t stream) {
  (void)in_sizes; (void)n_in; (void)out_size; (void)ws_size;
  const float* rgb = (const float*)d_in[0];
  const float* t   = (const float*)d_in[1];
  const float* wq1 = (const float*)d_in[2];
  const float* bq1 = (const float*)d_in[3];
  const float* wk1 = (const float*)d_in[4];
  const float* bk1 = (const float*)d_in[5];
  const float* wv1 = (const float*)d_in[6];
  const float* bv1 = (const float*)d_in[7];
  const float* wq2 = (const float*)d_in[8];
  const float* bq2 = (const float*)d_in[9];
  const float* wk2 = (const float*)d_in[10];
  const float* bk2 = (const float*)d_in[11];
  const float* wv2 = (const float*)d_in[12];
  const float* bv2 = (const float*)d_in[13];

  const size_t per = (size_t)2 * NB * NTOK * CCH;  // elements per qkv tensor
  _Float16* qbuf = (_Float16*)d_ws;
  _Float16* kbuf = qbuf + per;
  _Float16* vbuf = kbuf + per;
  _Float16* wbuf = vbuf + per;  // 6 * 256*256 f16

  dim3 blk(256);
  wcvt_kernel<<<dim3(32, 6), blk, 0, stream>>>(wq1, wk1, wv1, wq2, wk2, wv2,
                                               wbuf);
  proj_kernel<<<dim3(NTOK / TQ, 3, 2 * NB), blk, 0, stream>>>(
      rgb, t, bq1, bk1, bv1, bq2, bk2, bv2, wbuf, qbuf, kbuf, vbuf);
  attn_kernel<<<dim3(NTOK / TQ, 1, 2 * NB), blk, 0, stream>>>(
      rgb, t, qbuf, kbuf, vbuf, (float*)d_out);
}
